// BlockChunkedActivityRoutedNet_10788957847690
// MI455X (gfx1250) — matrix operand order repaired and checked by reading, not compiled, not run on hardware
//
#include <hip/hip_runtime.h>
#include <math.h>

// ---------------- problem constants ----------------
constexpr int NCHUNK = 8;
constexpr int TOPK   = 2;
constexpr int IN_F   = 4096;
constexpr int OUT_F  = 4096;
constexpr int BATCH  = 4096;
constexpr int CIN    = 512;   // IN_F / NCHUNK
constexpr int COUT   = 512;   // HID_F / NCHUNK
constexpr int KFIN   = TOPK * COUT; // 1024

// ---------------- GEMM tiling ----------------
constexpr int BM  = 128;
constexpr int BN  = 128;
constexpr int BK  = 32;
constexpr int LDT = 40;   // LDS pitch in bf16 elems: 80B rows -> 16B aligned, bank-conflict-free

typedef __bf16 bf16;
typedef __attribute__((ext_vector_type(16))) __bf16 v16bf;
typedef __attribute__((ext_vector_type(8)))  __bf16 v8bf;
typedef __attribute__((ext_vector_type(4)))  __bf16 v4bf;
typedef __attribute__((ext_vector_type(2)))  __bf16 v2bf;
typedef __attribute__((ext_vector_type(8)))  float  v8f;

union Frag16 { v16bf v; v8bf h[2]; };

// ---------------- kernel 0: zero activity accumulators ----------------
__global__ void zero_act_kernel(float* act) {
    if (threadIdx.x < NCHUNK) act[threadIdx.x] = 0.0f;
}

// ---------------- kernel 1: per-chunk sum(|x|) ----------------
// x viewed as (BATCH, NCHUNK, CIN); ordering by sum == ordering by mean.
__global__ __launch_bounds__(256) void activity_kernel(const float* __restrict__ x,
                                                       float* __restrict__ act) {
    const int t = threadIdx.x;
    float s[4] = {0.f, 0.f, 0.f, 0.f}; // thread t touches chunks c0,c0+2,c0+4,c0+6 (c0 = t>>7)
    for (int row = blockIdx.x; row < BATCH; row += gridDim.x) {
        const float4* xr = (const float4*)(x + (size_t)row * IN_F);
        #pragma unroll
        for (int i = 0; i < 4; ++i) {
            float4 v = xr[t + i * 256];
            s[i] += fabsf(v.x) + fabsf(v.y) + fabsf(v.z) + fabsf(v.w);
        }
    }
    __shared__ float red[NCHUNK];
    if (t < NCHUNK) red[t] = 0.f;
    __syncthreads();
    const int c0 = t >> 7;
    #pragma unroll
    for (int i = 0; i < 4; ++i) atomicAdd(&red[c0 + 2 * i], s[i]);
    __syncthreads();
    if (t < NCHUNK) atomicAdd(&act[t], red[t]);
}

// ---------------- kernel 2: top-2 (stable, matches jax.lax.top_k tie rule) ----------------
__global__ void topk_kernel(const float* __restrict__ act, int* __restrict__ topk) {
    if (threadIdx.x == 0) {
        int best = 0;
        for (int i = 1; i < NCHUNK; ++i) if (act[i] > act[best]) best = i;
        int second = (best == 0) ? 1 : 0;
        for (int i = 0; i < NCHUNK; ++i)
            if (i != best && act[i] > act[second]) second = i;
        topk[0] = best;
        topk[1] = second;
    }
}

// ---------------- LDS staging helpers ----------------
// A tile BM x BK, fp32 source -> bf16 LDS (row-major, pitch LDT)
__device__ __forceinline__ void stage_A_f32(bf16* As, const float* __restrict__ Ap,
                                            int lda, int k0, int t) {
    #pragma unroll
    for (int i = 0; i < 4; ++i) {
        int idx = t + i * 256;
        int r = idx >> 3, c4 = idx & 7;
        float4 v = *(const float4*)(Ap + (size_t)r * lda + k0 + c4 * 4);
        v4bf pv; pv[0] = (bf16)v.x; pv[1] = (bf16)v.y; pv[2] = (bf16)v.z; pv[3] = (bf16)v.w;
        *(v4bf*)(&As[r * LDT + c4 * 4]) = pv;
    }
}

// A tile BM x BK, bf16 source (straight 16B copies)
__device__ __forceinline__ void stage_A_bf16(bf16* As, const bf16* __restrict__ Ap,
                                             int lda, int k0, int t) {
    #pragma unroll
    for (int i = 0; i < 2; ++i) {
        int idx = t + i * 256;
        int r = idx >> 2, c = idx & 3;
        uint4 v = *(const uint4*)(Ap + (size_t)r * lda + k0 + c * 8);
        *(uint4*)(&As[r * LDT + c * 8]) = v;
    }
}

// B tile: fp32 row-major (K x N) source; store K-transposed Bs[n][k].
// Each thread handles a (k, k+1) row pair so every LDS store is a packed
// bf16x2 b32 store (cvt_pk_bf16_f32 pairs along K; kr even -> 4B aligned).
__device__ __forceinline__ void stage_B_f32t(bf16* Bs, const float* __restrict__ Bp,
                                             int ldb, int k0, int n0, int t) {
    #pragma unroll
    for (int i = 0; i < 2; ++i) {
        int idx = t + i * 256;           // 512 items: 16 k-pairs x 32 float4 cols
        int kp = idx >> 5;               // k rows 2kp, 2kp+1
        int q  = idx & 31;
        const float* p0 = Bp + (size_t)(k0 + 2 * kp) * ldb + n0 + q * 4;
        float4 a = *(const float4*)(p0);
        float4 b = *(const float4*)(p0 + ldb);
        v2bf w0; w0[0] = (bf16)a.x; w0[1] = (bf16)b.x;
        v2bf w1; w1[0] = (bf16)a.y; w1[1] = (bf16)b.y;
        v2bf w2; w2[0] = (bf16)a.z; w2[1] = (bf16)b.z;
        v2bf w3; w3[0] = (bf16)a.w; w3[1] = (bf16)b.w;
        *(v2bf*)(&Bs[(q * 4 + 0) * LDT + 2 * kp]) = w0;
        *(v2bf*)(&Bs[(q * 4 + 1) * LDT + 2 * kp]) = w1;
        *(v2bf*)(&Bs[(q * 4 + 2) * LDT + 2 * kp]) = w2;
        *(v2bf*)(&Bs[(q * 4 + 3) * LDT + 2 * kp]) = w3;
    }
}

// ---------------- shared WMMA tile compute ----------------
// As: BM x BK (row-major, pitch LDT), Bs: BN x BK (K-transposed, pitch LDT)
__device__ __forceinline__ void wmma_tile(const bf16* As, const bf16* Bs,
                                          int wm, int wn, int lane,
                                          v8f acc[2][4]) {
    const int hf = lane >> 4;   // lane half selects K sub-range per ISA layout
    const int lm = lane & 15;
    Frag16 af[2], bfr[4];
    #pragma unroll
    for (int i = 0; i < 2; ++i) {
        const bf16* p = &As[(wm * 32 + i * 16 + lm) * LDT + hf * 8];
        af[i].h[0] = *(const v8bf*)(p);        // K = 8*hf .. +7
        af[i].h[1] = *(const v8bf*)(p + 16);   // K = 16 + 8*hf .. +7
    }
    #pragma unroll
    for (int j = 0; j < 4; ++j) {
        const bf16* p = &Bs[(wn * 64 + j * 16 + lm) * LDT + hf * 8];
        bfr[j].h[0] = *(const v8bf*)(p);       // K = 8*hf .. +7
        bfr[j].h[1] = *(const v8bf*)(p + 16);  // K = 16 + 8*hf .. +7
    }
    #pragma unroll
    for (int i = 0; i < 2; ++i)
        #pragma unroll
        for (int j = 0; j < 4; ++j)
            acc[i][j] = __builtin_amdgcn_wmma_f32_16x16x32_bf16(
                false, af[i].v, false, bfr[j].v, (short)0, acc[i][j], false, false);
}

// ---------------- kernel 3: routed chunk GEMM ----------------
// H[:, k*512 : (k+1)*512] = x[:, chunk*512:(chunk+1)*512] @ W_chunks[chunk] + b_chunks[chunk]
__global__ __launch_bounds__(256) void chunk_gemm_kernel(
    const float* __restrict__ x, const float* __restrict__ Wc,
    const float* __restrict__ bc, const int* __restrict__ topk,
    bf16* __restrict__ H) {
    __shared__ __align__(16) bf16 As[2][BM * LDT];
    __shared__ __align__(16) bf16 Bs[2][BN * LDT];

    const int t = threadIdx.x;
    const int lane = t & 31, wave = t >> 5;
    const int wm = wave & 3, wn = wave >> 2;
    const int kSel  = blockIdx.z;
    const int chunk = topk[kSel];
    const int n0 = blockIdx.x * BN;
    const int m0 = blockIdx.y * BM;

    v8f acc[2][4];
    #pragma unroll
    for (int i = 0; i < 2; ++i)
        #pragma unroll
        for (int j = 0; j < 4; ++j)
            #pragma unroll
            for (int r = 0; r < 8; ++r) acc[i][j][r] = 0.0f;

    const float* Ap = x  + (size_t)m0 * IN_F + (size_t)chunk * CIN;
    const float* Bp = Wc + (size_t)chunk * CIN * COUT;

    constexpr int NIT = CIN / BK; // 16
    stage_A_f32 (As[0], Ap, IN_F, 0, t);
    stage_B_f32t(Bs[0], Bp, COUT, 0, n0, t);
    __syncthreads();

    for (int it = 0; it < NIT; ++it) {
        const int buf = it & 1;
        const int k0  = it * BK;
        if (it + 1 < NIT) {
            stage_A_f32 (As[buf ^ 1], Ap, IN_F, k0 + BK, t);
            stage_B_f32t(Bs[buf ^ 1], Bp, COUT, k0 + BK, n0, t);
        }
        if (it + 2 < NIT) {
            // run the L0/L2 prefetch path two tiles ahead (global_prefetch_b8)
            __builtin_prefetch(Ap + (size_t)(t >> 1) * IN_F + k0 + 2 * BK + (t & 1) * 16, 0, 3);
            __builtin_prefetch(Bp + (size_t)(k0 + 2 * BK + (t >> 3)) * COUT + n0 + (t & 7) * 16, 0, 3);
        }
        wmma_tile(As[buf], Bs[buf], wm, wn, lane, acc);
        __syncthreads();
    }

    const int hf = lane >> 4, lm = lane & 15;
    #pragma unroll
    for (int i = 0; i < 2; ++i) {
        #pragma unroll
        for (int j = 0; j < 4; ++j) {
            int col = n0 + wn * 64 + j * 16 + lm;
            float bias = bc[chunk * COUT + col];
            #pragma unroll
            for (int r = 0; r < 8; ++r) {
                int row = m0 + wm * 32 + i * 16 + hf * 8 + r;
                H[(size_t)row * KFIN + kSel * COUT + col] = (bf16)(acc[i][j][r] + bias);
            }
        }
    }
}

// ---------------- kernel 4: final GEMM: out = H @ W_final + b_final ----------------
__global__ __launch_bounds__(256) void final_gemm_kernel(
    const bf16* __restrict__ H, const float* __restrict__ Wf,
    const float* __restrict__ bfin, float* __restrict__ out) {
    __shared__ __align__(16) bf16 As[2][BM * LDT];
    __shared__ __align__(16) bf16 Bs[2][BN * LDT];

    const int t = threadIdx.x;
    const int lane = t & 31, wave = t >> 5;
    const int wm = wave & 3, wn = wave >> 2;
    const int n0 = blockIdx.x * BN;
    const int m0 = blockIdx.y * BM;

    v8f acc[2][4];
    #pragma unroll
    for (int i = 0; i < 2; ++i)
        #pragma unroll
        for (int j = 0; j < 4; ++j)
            #pragma unroll
            for (int r = 0; r < 8; ++r) acc[i][j][r] = 0.0f;

    const bf16* Ap = H + (size_t)m0 * KFIN;

    constexpr int NIT = KFIN / BK; // 32
    stage_A_bf16(As[0], Ap, KFIN, 0, t);
    stage_B_f32t(Bs[0], Wf, OUT_F, 0, n0, t);
    __syncthreads();

    for (int it = 0; it < NIT; ++it) {
        const int buf = it & 1;
        const int k0  = it * BK;
        if (it + 1 < NIT) {
            stage_A_bf16(As[buf ^ 1], Ap, KFIN, k0 + BK, t);
            stage_B_f32t(Bs[buf ^ 1], Wf, OUT_F, k0 + BK, n0, t);
        }
        if (it + 2 < NIT) {
            __builtin_prefetch(Ap + (size_t)(t >> 1) * KFIN + k0 + 2 * BK + (t & 1) * 16, 0, 3);
            __builtin_prefetch(Wf + (size_t)(k0 + 2 * BK + (t >> 3)) * OUT_F + n0 + (t & 7) * 16, 0, 3);
        }
        wmma_tile(As[buf], Bs[buf], wm, wn, lane, acc);
        __syncthreads();
    }

    const int hf = lane >> 4, lm = lane & 15;
    #pragma unroll
    for (int i = 0; i < 2; ++i) {
        #pragma unroll
        for (int j = 0; j < 4; ++j) {
            int col = n0 + wn * 64 + j * 16 + lm;
            float bias = bfin[col];
            #pragma unroll
            for (int r = 0; r < 8; ++r) {
                int row = m0 + wm * 32 + i * 16 + hf * 8 + r;
                out[(size_t)row * OUT_F + col] = acc[i][j][r] + bias;
            }
        }
    }
}

// ---------------- launcher ----------------
extern "C" void kernel_launch(void* const* d_in, const int* in_sizes, int n_in,
                              void* d_out, int out_size, void* d_ws, size_t ws_size,
                              hipStream_t stream) {
    const float* x   = (const float*)d_in[0];
    const float* Wc  = (const float*)d_in[1];
    const float* bc  = (const float*)d_in[2];
    const float* Wf  = (const float*)d_in[3];
    const float* bfv = (const float*)d_in[4];
    float* out = (float*)d_out;

    char*  ws   = (char*)d_ws;
    float* act  = (float*)ws;              // 8 floats
    int*   topk = (int*)(ws + 64);         // 2 ints
    bf16*  H    = (bf16*)(ws + 256);       // 4096 x 1024 bf16 (~8 MB)

    zero_act_kernel<<<1, 32, 0, stream>>>(act);
    activity_kernel<<<512, 256, 0, stream>>>(x, act);
    topk_kernel<<<1, 32, 0, stream>>>(act, topk);

    dim3 g3(COUT / BN, BATCH / BM, TOPK);  // (4, 32, 2)
    chunk_gemm_kernel<<<g3, 256, 0, stream>>>(x, Wc, bc, topk, H);

    dim3 g4(OUT_F / BN, BATCH / BM, 1);    // (32, 32)
    final_gemm_kernel<<<g4, 256, 0, stream>>>(H, Wf, bfv, out);
}
